// CharLSTM_15513421873237
// MI455X (gfx1250) — compile-verified, matrix operand order checked by reading, not run on hardware
//
#include <hip/hip_runtime.h>
#include <hip/hip_bf16.h>
#include <stdint.h>

// CharLSTM on gfx1250 (MI455X), wave32 + WMMA:
//  - pack kernels: emb gather -> bf16 X; weights -> bf16 in exact WMMA
//    B-fragment lane layout (ISA 7.12.2); biases gathered.
//  - pregemm: Pre[g][t][n] = X @ Wx_g + b_g via v_wmma_f32_16x16x32_bf16
//    (true GEMM over time, no recurrence; M-tiles of 16 over T=1024).
//  - persistent lstm kernel (64 blocks x 8 waves): block b owns h-columns
//    [32b,32b+32). Its 8 waves compute the 4 gates x 2 col-tiles for those
//    columns (h @ Wh, A = h replicated across rows), activations go to LDS,
//    the c/h update is block-local (c lives in registers), and ONE grid
//    barrier per step publishes the new bf16 h. Logit tiles (h @ Wout) for
//    step t-1 overlap on wave 7 of blocks 0..15. All bf16 weights (~43MB)
//    stay resident in the 192MB L2. The h broadcast into LDS uses the
//    ASYNCcnt-tracked global_load_async_to_lds_b128 path.

#define T_LEN 1024
#define NCH   256
#define EDIM  512
#define HDIM  2048
#define NBLK  64
#define NTHR  256

typedef __attribute__((ext_vector_type(16))) __bf16 v16bf;
typedef __attribute__((ext_vector_type(8)))  __bf16 v8bf;
typedef __attribute__((ext_vector_type(8)))  float  v8f;
typedef __attribute__((ext_vector_type(4)))  int    v4i;

// address-space qualified 128-bit vector types for the async-LDS builtins
typedef v4i __attribute__((address_space(1))) v4i_g;   // global
typedef v4i __attribute__((address_space(3))) v4i_l;   // LDS

#if defined(__HIP_DEVICE_COMPILE__)
#if __has_builtin(__builtin_amdgcn_global_load_async_to_lds_b128) && \
    __has_builtin(__builtin_amdgcn_s_wait_asynccnt)
#define USE_ASYNC_LDS 1
#endif
#endif

union Frag16 { v16bf v; struct { v8bf lo; v8bf hi; } p; };

// A/B 16-bit fragment halves: lanes 0-15 hold K {0..7,16..23}, lanes 16-31
// hold K {8..15,24..31} within the 32-wide chunk. base already includes the
// lane's +0/+8 K offset.
__device__ __forceinline__ v16bf load_frag_split(const __bf16* base) {
  Frag16 f;
  f.p.lo = *(const v8bf*)(base);
  f.p.hi = *(const v8bf*)(base + 16);
  return f.v;
}

#define WMMA_BF16(A, B, C) \
  __builtin_amdgcn_wmma_f32_16x16x32_bf16(false, (A), false, (B), (short)0, (C), false, false)

// ---------------- packing kernels ----------------

__global__ void pack_x_kernel(const int* __restrict__ seq,
                              const float* __restrict__ emb,
                              __bf16* __restrict__ XB) {
  int i = blockIdx.x * blockDim.x + threadIdx.x;
  if (i >= T_LEN * EDIM) return;
  int t = i >> 9;           // / EDIM
  int e = i & (EDIM - 1);
  XB[i] = (__bf16)emb[(size_t)seq[t] * EDIM + e];
}

// Pack W[K][N] (row-major, ldn = N, rows starting at rowOff) into per-wave B
// fragments: group gid = (ct*KC + kc)*32 + lane, 16 bf16 per group.
__global__ void pack_w_kernel(const float* __restrict__ W, int rowOff, int K, int ldn,
                              int nGroups, __bf16* __restrict__ dst) {
  int gid = blockIdx.x * blockDim.x + threadIdx.x;
  if (gid >= nGroups) return;
  int KC   = K >> 5;
  int lane = gid & 31;
  int kc   = (gid >> 5) % KC;
  int ct   = (gid >> 5) / KC;
  int col  = (ct << 4) + (lane & 15);
  int kb   = (kc << 5) + ((lane & 16) ? 8 : 0);
  __bf16* o = dst + (size_t)gid * 16;
#pragma unroll
  for (int j = 0; j < 8; ++j)
    o[j] = (__bf16)W[(size_t)(rowOff + kb + j) * ldn + col];
#pragma unroll
  for (int j = 0; j < 8; ++j)
    o[8 + j] = (__bf16)W[(size_t)(rowOff + kb + 16 + j) * ldn + col];
}

__global__ void pack_bias_kernel(const float* __restrict__ b0, const float* __restrict__ b1,
                                 const float* __restrict__ b2, const float* __restrict__ b3,
                                 float* __restrict__ BIASG) {
  int i = blockIdx.x * blockDim.x + threadIdx.x;
  if (i >= 4 * HDIM) return;
  int g = i >> 11;
  int n = i & (HDIM - 1);
  const float* s = (g == 0) ? b0 : (g == 1) ? b1 : (g == 2) ? b2 : b3;
  BIASG[i] = s[n];
}

// ---------------- pre-GEMM over time: Pre = X @ Wx + b ----------------

__global__ void __launch_bounds__(NTHR) pregemm_kernel(const __bf16* __restrict__ XB,
                                                       const __bf16* __restrict__ WXP,
                                                       const float* __restrict__ BIASG,
                                                       float* __restrict__ PRE) {
  int lane = threadIdx.x & 31;
  int wave = (blockIdx.x * NTHR + threadIdx.x) >> 5;   // 0..32767
  int mt  = wave >> 9;        // T tile (64)
  int nt  = wave & 511;       // gate-major col tile (512)
  int g   = nt >> 7;
  int ctg = nt & 127;
  const __bf16* bbase = WXP + ((size_t)g * 128 + ctg) * (size_t)(16 * 512);
  int row  = (mt << 4) + (lane & 15);
  int koff = (lane & 16) ? 8 : 0;
  const __bf16* arow = XB + (size_t)row * EDIM + koff;

  v8f acc0 = {}, acc1 = {};
#pragma unroll 4
  for (int kc = 0; kc < 16; kc += 2) {
    v16bf a0 = load_frag_split(arow + (kc << 5));
    v16bf b0 = *(const v16bf*)(bbase + ((size_t)kc * 32 + lane) * 16);
    acc0 = WMMA_BF16(a0, b0, acc0);
    v16bf a1 = load_frag_split(arow + ((kc + 1) << 5));
    v16bf b1 = *(const v16bf*)(bbase + ((size_t)(kc + 1) * 32 + lane) * 16);
    acc1 = WMMA_BF16(a1, b1, acc1);
  }
  v8f acc = acc0 + acc1;
  int n = (ctg << 4) + (lane & 15);
  float bias = BIASG[(g << 11) + n];
  int m0 = (lane & 16) ? 8 : 0;
  float* base = PRE + ((size_t)g * T_LEN + (size_t)(mt << 4) + m0) * HDIM + n;
#pragma unroll
  for (int j = 0; j < 8; ++j)
    base[(size_t)j * HDIM] = acc[j] + bias;
}

// ---------------- persistent recurrent kernel ----------------

__device__ __forceinline__ void grid_barrier(int* cnt, int* gen) {
  __syncthreads();
  if (threadIdx.x == 0) {
    __threadfence();
    int g = __hip_atomic_load(gen, __ATOMIC_RELAXED, __HIP_MEMORY_SCOPE_AGENT);
    int arrived = __hip_atomic_fetch_add(cnt, 1, __ATOMIC_ACQ_REL, __HIP_MEMORY_SCOPE_AGENT);
    if (arrived == NBLK - 1) {
      __hip_atomic_store(cnt, 0, __ATOMIC_RELAXED, __HIP_MEMORY_SCOPE_AGENT);
      __hip_atomic_store(gen, g + 1, __ATOMIC_RELEASE, __HIP_MEMORY_SCOPE_AGENT);
    } else {
      while (__hip_atomic_load(gen, __ATOMIC_ACQUIRE, __HIP_MEMORY_SCOPE_AGENT) == g)
        __builtin_amdgcn_s_sleep(2);
    }
    __threadfence();
  }
  __syncthreads();
}

// One 16-col logit tile: dst[lane] = (h @ Wout)[tile] + bout[tile]
__device__ __forceinline__ void logits_tile(const __bf16* lds_h, const __bf16* wob,
                                            const float* boutp, float* dst,
                                            int lane, int koff) {
  v8f l0 = {}, l1 = {};
#pragma unroll 4
  for (int kc = 0; kc < 64; kc += 2) {
    v16bf a0 = load_frag_split(lds_h + (kc << 5) + koff);
    v16bf b0 = *(const v16bf*)(wob + ((size_t)kc * 32 + lane) * 16);
    l0 = WMMA_BF16(a0, b0, l0);
    v16bf a1 = load_frag_split(lds_h + ((kc + 1) << 5) + koff);
    v16bf b1 = *(const v16bf*)(wob + ((size_t)(kc + 1) * 32 + lane) * 16);
    l1 = WMMA_BF16(a1, b1, l1);
  }
  v8f ls = l0 + l1;
  if (lane < 16) dst[lane] = ls[0] + boutp[lane];
}

__global__ void __launch_bounds__(NTHR) lstm_kernel(const float* __restrict__ PRE,
                                                    const __bf16* __restrict__ WHP,
                                                    const __bf16* __restrict__ WOP,
                                                    const float* __restrict__ boutp,
                                                    __bf16* __restrict__ hBF,
                                                    int* __restrict__ bar,
                                                    float* __restrict__ out) {
  __shared__ __bf16 lds_h[HDIM];
  __shared__ float  gact[4][32];
  const int tid  = threadIdx.x;
  const int lane = tid & 31;
  const int w    = tid >> 5;                     // wave in block: 0..7
  const int g    = w >> 1;                       // gate 0..3
  const int sub  = w & 1;                        // col-tile half
  const int ct   = (blockIdx.x << 1) + sub;      // col tile within gate 0..127
  const int koff = (lane & 16) ? 8 : 0;
  const int nloc = (ct << 4) + (lane & 15);      // column within gate
  const __bf16* whb = WHP + (size_t)(g * 128 + ct) * (size_t)(64 * 512);
  const bool doLog = (w == 7) && (blockIdx.x < 16);
  const __bf16* wob = WOP + (size_t)blockIdx.x * (size_t)(64 * 512);
  const int colBase = blockIdx.x << 5;           // this block's 32 h-columns
  int* cnt = bar;
  int* gen = bar + 1;
  float c_reg = 0.0f;                            // cell state, block-private

  for (int t = 0; t < T_LEN; ++t) {
    { // broadcast h (2048 bf16 = 4KB) into LDS, 16B per thread
      int i = tid << 3;
#ifdef USE_ASYNC_LDS
      __builtin_amdgcn_global_load_async_to_lds_b128(
          (v4i_g*)(uintptr_t)(hBF + i),
          (v4i_l*)(uint32_t)(uintptr_t)(lds_h + i), 0, 0);
      __builtin_amdgcn_s_wait_asynccnt(0);
#else
      *(v8bf*)(lds_h + i) = *(const v8bf*)(hBF + i);
#endif
    }
    __syncthreads();

    // gate tile: y[16 cols] = h @ Wh_g ; 4 independent accumulator chains
    v8f a0 = {}, a1 = {}, a2 = {}, a3 = {};
#pragma unroll 4
    for (int kc = 0; kc < 64; kc += 4) {
      if ((kc & 15) == 0 && kc + 16 < 64)
        __builtin_prefetch(whb + ((size_t)(kc + 16) * 32 + lane) * 16, 0, 1);
      v16bf fa0 = load_frag_split(lds_h + ((kc + 0) << 5) + koff);
      v16bf fb0 = *(const v16bf*)(whb + ((size_t)(kc + 0) * 32 + lane) * 16);
      a0 = WMMA_BF16(fa0, fb0, a0);
      v16bf fa1 = load_frag_split(lds_h + ((kc + 1) << 5) + koff);
      v16bf fb1 = *(const v16bf*)(whb + ((size_t)(kc + 1) * 32 + lane) * 16);
      a1 = WMMA_BF16(fa1, fb1, a1);
      v16bf fa2 = load_frag_split(lds_h + ((kc + 2) << 5) + koff);
      v16bf fb2 = *(const v16bf*)(whb + ((size_t)(kc + 2) * 32 + lane) * 16);
      a2 = WMMA_BF16(fa2, fb2, a2);
      v16bf fa3 = load_frag_split(lds_h + ((kc + 3) << 5) + koff);
      v16bf fb3 = *(const v16bf*)(whb + ((size_t)(kc + 3) * 32 + lane) * 16);
      a3 = WMMA_BF16(fa3, fb3, a3);
    }
    v8f acc = (a0 + a1) + (a2 + a3);
    if (lane < 16) {
      float u = PRE[((size_t)g * T_LEN + t) * HDIM + nloc] + acc[0];
      float v = (g == 3) ? tanhf(u) : (1.0f / (1.0f + __expf(-u)));
      gact[g][(sub << 4) + (lane & 15)] = v;
    }

    // logits for step t-1 (same h as this step's gates); wave 7, blocks 0..15
    if (doLog && t > 0)
      logits_tile(lds_h, wob, boutp + (blockIdx.x << 4),
                  out + (size_t)(t - 1) * NCH + (blockIdx.x << 4), lane, koff);

    __syncthreads();   // gact ready (block-local)

    // block-local cell/hidden update for this block's 32 columns
    if (tid < 32) {
      float fg = gact[0][tid];
      float ig = gact[1][tid];
      float og = gact[2][tid];
      float gg = gact[3][tid];
      float cn = fg * c_reg + ig * gg;
      float hn = og * tanhf(cn);
      c_reg = cn;
      int j = colBase + tid;
      hBF[j] = (__bf16)hn;
      if (t == T_LEN - 1) {
        out[(size_t)T_LEN * NCH + j] = hn;          // h_last
        out[(size_t)T_LEN * NCH + HDIM + j] = cn;   // c_last
      }
    }

    grid_barrier(cnt, gen);   // publish new h to all blocks (1 per step)
  }

  // final logits out[T-1] with h_T
  if (blockIdx.x < 16) {
    int i = tid << 3;
    *(v8bf*)(lds_h + i) = *(const v8bf*)(hBF + i);
    __syncthreads();
    if (w == 7)
      logits_tile(lds_h, wob, boutp + (blockIdx.x << 4),
                  out + (size_t)(T_LEN - 1) * NCH + (blockIdx.x << 4), lane, koff);
  }
}

// ---------------- host launch ----------------

extern "C" void kernel_launch(void* const* d_in, const int* in_sizes, int n_in,
                              void* d_out, int out_size, void* d_ws, size_t ws_size,
                              hipStream_t stream) {
  (void)in_sizes; (void)n_in; (void)out_size; (void)ws_size;
  const int*   seq  = (const int*)  d_in[0];
  const float* emb  = (const float*)d_in[1];
  const float* Wf   = (const float*)d_in[2];
  const float* bf_  = (const float*)d_in[3];
  const float* Wi   = (const float*)d_in[4];
  const float* bi_  = (const float*)d_in[5];
  const float* Wo   = (const float*)d_in[6];
  const float* bo_  = (const float*)d_in[7];
  const float* Wc   = (const float*)d_in[8];
  const float* bc_  = (const float*)d_in[9];
  const float* Wout = (const float*)d_in[10];
  const float* boutp= (const float*)d_in[11];
  float* out = (float*)d_out;
  char*  ws  = (char*)d_ws;

  size_t off = 0;
  auto take = [&](size_t bytes) { size_t o = off; off += (bytes + 255) & ~(size_t)255; return o; };
  __bf16* XB    = (__bf16*)(ws + take((size_t)T_LEN * EDIM * 2));
  __bf16* WXP   = (__bf16*)(ws + take((size_t)4 * EDIM * HDIM * 2));
  __bf16* WHP   = (__bf16*)(ws + take((size_t)4 * HDIM * HDIM * 2));
  __bf16* WOP   = (__bf16*)(ws + take((size_t)HDIM * NCH * 2));
  float*  PRE   = (float*) (ws + take((size_t)4 * T_LEN * HDIM * 4));
  float*  BIASG = (float*) (ws + take((size_t)4 * HDIM * 4));
  size_t stateOff = off;
  __bf16* hBF   = (__bf16*)(ws + take((size_t)HDIM * 2));
  int*    bar   = (int*)   (ws + take(256));
  size_t stateBytes = off - stateOff;

  // zero h and barrier state every launch (deterministic graph replay)
  (void)hipMemsetAsync(ws + stateOff, 0, stateBytes, stream);

  pack_x_kernel<<<(T_LEN * EDIM + NTHR - 1) / NTHR, NTHR, 0, stream>>>(seq, emb, XB);

  const float* Wg[4] = {Wf, Wi, Wo, Wc};
  const int gx = (HDIM / 16) * (EDIM / 32) * 32;   // 65536 groups per gate (Wx)
  const int gh = (HDIM / 16) * (HDIM / 32) * 32;   // 262144 groups per gate (Wh)
  for (int g = 0; g < 4; ++g) {
    pack_w_kernel<<<gx / NTHR, NTHR, 0, stream>>>(Wg[g], 0,    EDIM, HDIM, gx,
                                                  WXP + (size_t)g * gx * 16);
    pack_w_kernel<<<gh / NTHR, NTHR, 0, stream>>>(Wg[g], EDIM, HDIM, HDIM, gh,
                                                  WHP + (size_t)g * gh * 16);
  }
  const int go = (NCH / 16) * (HDIM / 32) * 32;    // 32768 groups (Wout)
  pack_w_kernel<<<go / NTHR, NTHR, 0, stream>>>(Wout, 0, HDIM, NCH, go, WOP);
  pack_bias_kernel<<<(4 * HDIM + NTHR - 1) / NTHR, NTHR, 0, stream>>>(bf_, bi_, bo_, bc_, BIASG);

  pregemm_kernel<<<(T_LEN / 16) * (4 * HDIM / 16) / 8, NTHR, 0, stream>>>(XB, WXP, BIASG, PRE);
  lstm_kernel<<<NBLK, NTHR, 0, stream>>>(PRE, WHP, WOP, boutp, hBF, bar, out);
}
